// GloVe_Refiner_14955076124735
// MI455X (gfx1250) — matrix operand order repaired and checked by reading, not compiled, not run on hardware
//
#include <hip/hip_runtime.h>
#include <hip/hip_fp16.h>
#include <stdint.h>

#define N_TOK 50176
#define M_ATTR 312
#define M_PAD 320
#define DIM 768
#define HID 384
#define KSPLIT 16
#define KCHUNK 3136   // N_TOK / KSPLIT
#define MOM 0.8f
#define TOKB 256      // tokens per block in gemm1

#if defined(__HIP_DEVICE_COMPILE__) && \
    __has_builtin(__builtin_amdgcn_global_load_async_to_lds_b128) && \
    __has_builtin(__builtin_amdgcn_s_wait_asynccnt)
#define HAVE_ASYNC 1
#else
#define HAVE_ASYNC 0
#endif

#if HAVE_ASYNC
#define G1STR 40   // halves per LDS row: 80B rows -> 16B aligned + conflict-free (20 dwords)
typedef int v4i __attribute__((vector_size(16)));
typedef __attribute__((address_space(1))) v4i gl_v4i;   // prints as "__device__" in diags
typedef __attribute__((address_space(3))) v4i ds_v4i;
__device__ __forceinline__ void async_b128(const void* g, void* l) {
    __builtin_amdgcn_global_load_async_to_lds_b128(
        (gl_v4i*)(uintptr_t)g,
        (ds_v4i*)(unsigned)(uintptr_t)l,   // low 32 bits of generic LDS ptr == LDS offset
        0, 0);
}
#else
#define G1STR 34
#endif

typedef __attribute__((ext_vector_type(16))) _Float16 v16h;
typedef __attribute__((ext_vector_type(8)))  float    v8f;

union Frag { unsigned u[8]; v16h h; };

__device__ __forceinline__ float block_sum(float v, float* red) {
    int t = threadIdx.x;
    red[t] = v; __syncthreads();
    for (int s = 128; s; s >>= 1) { if (t < s) red[t] += red[t + s]; __syncthreads(); }
    float r = red[0]; __syncthreads();
    return r;
}

// ---------- K0: normalize glove rows -> ng (f32) + ng_h (f16, padded to 320 rows); zero `has`
__global__ void k_prep_glove(const float* __restrict__ glove, float* __restrict__ ng,
                             _Float16* __restrict__ ng_h, int* __restrict__ has) {
    int m = blockIdx.x, t = threadIdx.x;
    __shared__ float red[256];
    if (t == 0) has[m] = 0;
    if (m >= M_ATTR) {
        for (int d = t; d < DIM; d += 256) ng_h[(size_t)m * DIM + d] = (_Float16)0.f;
        return;
    }
    float x[3], ss = 0.f;
    for (int i = 0; i < 3; i++) { x[i] = glove[(size_t)m * DIM + t + 256 * i]; ss += x[i] * x[i]; }
    ss = block_sum(ss, red);
    float inv = 1.0f / fmaxf(sqrtf(ss), 1e-12f);
    for (int i = 0; i < 3; i++) {
        float v = x[i] * inv;
        ng[(size_t)m * DIM + t + 256 * i] = v;
        ng_h[(size_t)m * DIM + t + 256 * i] = (_Float16)v;
    }
}

// ---------- K1: normalize tokens -> nlf_h (f16), store recovery norm M_j = max(||lf||, 1e-12)
__global__ void k_prep_tokens(const float* __restrict__ lt, _Float16* __restrict__ nlf,
                              float* __restrict__ normM) {
    int j = blockIdx.x, t = threadIdx.x;
    __shared__ float red[256];
    float x[3], ss = 0.f;
    for (int i = 0; i < 3; i++) { x[i] = lt[(size_t)j * DIM + t + 256 * i]; ss += x[i] * x[i]; }
    ss = block_sum(ss, red);
    float Mj = fmaxf(sqrtf(ss), 1e-12f);
    float inv = 1.0f / Mj;
    for (int i = 0; i < 3; i++)
        nlf[(size_t)j * DIM + t + 256 * i] = (_Float16)(x[i] * inv);
    if (t == 0) normM[j] = Mj;
}

// ---------- GEMM1 fragment compute: 2 j-frags per wave
__device__ __forceinline__ void g1_compute(const _Float16* sAb, const _Float16* sBb,
                                           v8f* acc, int lane, int w) {
    Frag a;
    int Mr = lane & 15, kb = (lane < 16) ? 0 : 16;
#pragma unroll
    for (int v = 0; v < 8; v++) {
        int K = kb + 2 * v;
        a.u[v] = *(const unsigned*)&sAb[Mr * G1STR + K];
    }
    int kbB = (lane < 16) ? 0 : 16;
#pragma unroll
    for (int jb = 0; jb < 2; jb++) {
        Frag b;
        int jl = w * 32 + jb * 16 + (lane & 15);
#pragma unroll
        for (int v = 0; v < 8; v++)
            b.u[v] = *(const unsigned*)&sBb[jl * G1STR + kbB + 2 * v];
        acc[jb] = __builtin_amdgcn_wmma_f32_16x16x32_f16(
            false, a.h, false, b.h, (short)0, acc[jb], false, false);
    }
}

#if HAVE_ASYNC
__device__ __forceinline__ void g1_issueA(const _Float16* ng_h, _Float16* sAbuf,
                                          int m0, int k0, int t) {
    if (t < 64) {   // 16 rows x 4 chunks of 16B
        int r = t >> 2, ch = (t & 3) * 16;
        async_b128((const char*)(ng_h + (size_t)(m0 + r) * DIM + k0) + ch,
                   (char*)(sAbuf + r * G1STR) + ch);
    }
}
__device__ __forceinline__ void g1_issueB(const _Float16* nlf, _Float16* sBbuf,
                                          int jblk, int k0, int t) {
#pragma unroll
    for (int i = 0; i < 4; i++) {   // 256 rows x 4 chunks = 1024, 4 per thread
        int idx = t + 256 * i;
        int jl = idx >> 2, ch = (idx & 3) * 16;
        async_b128((const char*)(nlf + (size_t)(jblk + jl) * DIM + k0) + ch,
                   (char*)(sBbuf + jl * G1STR) + ch);
    }
}
#endif

// ---------- K2: score[320 x N] = ng_h @ nlf_h^T, f16 WMMA f32 accum.
// Block: 16 M-rows x 256 tokens; K streamed in 32-slabs; async double-buffered LDS.
__global__ __launch_bounds__(256) void k_gemm1(const _Float16* __restrict__ ng_h,
                                               const _Float16* __restrict__ nlf,
                                               float* __restrict__ score) {
    int t = threadIdx.x, lane = t & 31, w = t >> 5;
    int m0 = blockIdx.y * 16;
    int jblk = blockIdx.x * TOKB;
    v8f acc[2] = {};

#if HAVE_ASYNC
    __shared__ __align__(16) _Float16 sA[2][16 * G1STR];
    __shared__ __align__(16) _Float16 sB[2][TOKB * G1STR];
    g1_issueA(ng_h, sA[0], m0, 0, t);
    g1_issueB(nlf, sB[0], jblk, 0, t);
    __builtin_amdgcn_s_wait_asynccnt(0);
    __syncthreads();
    for (int k0 = 0; k0 < DIM; k0 += 32) {
        int cur = (k0 >> 5) & 1;
        if (k0 + 32 < DIM) {            // prefetch next slab into other buffer
            g1_issueA(ng_h, sA[cur ^ 1], m0, k0 + 32, t);
            g1_issueB(nlf, sB[cur ^ 1], jblk, k0 + 32, t);
        }
        g1_compute(sA[cur], sB[cur], acc, lane, w);   // overlaps with async DMA
        __builtin_amdgcn_s_wait_asynccnt(0);
        __syncthreads();
    }
#else
    __shared__ __align__(16) _Float16 sA[1][16 * G1STR];
    __shared__ __align__(16) _Float16 sB[1][TOKB * G1STR];
    for (int k0 = 0; k0 < DIM; k0 += 32) {
        { int r = t >> 4, c2 = (t & 15) * 2;
          *(unsigned*)&sA[0][r * G1STR + c2] =
              *(const unsigned*)&ng_h[(size_t)(m0 + r) * DIM + k0 + c2]; }
        for (int idx = t; idx < TOKB * 16; idx += 256) {
            int jl = idx >> 4, c2 = (idx & 15) * 2;
            *(unsigned*)&sB[0][jl * G1STR + c2] =
                *(const unsigned*)&nlf[(size_t)(jblk + jl) * DIM + k0 + c2];
        }
        __syncthreads();
        g1_compute(sA[0], sB[0], acc, lane, w);
        __syncthreads();
    }
#endif
    int rbase = m0 + ((lane < 16) ? 0 : 8);
    int cbase = jblk + w * 32 + (lane & 15);
#pragma unroll
    for (int jb = 0; jb < 2; jb++)
#pragma unroll
        for (int r = 0; r < 8; r++)
            score[(size_t)(rbase + r) * N_TOK + cbase + jb * 16] = acc[jb][r];
}

// ---------- K3a: per-row online softmax stats (max, sum)
__global__ void k_row_stats(const float* __restrict__ score, float* __restrict__ rowmax,
                            float* __restrict__ rowsum) {
    int i = blockIdx.x, t = threadIdx.x;
    __shared__ float smax[256], ssum[256];
    const float* row = score + (size_t)i * N_TOK;
    float mx = -3.0e38f, sm = 0.f;
    for (int j = t; j < N_TOK; j += 256) {
        float v = row[j];
        if (v > mx) { sm = sm * expf(mx - v) + 1.f; mx = v; }
        else sm += expf(v - mx);
    }
    smax[t] = mx; ssum[t] = sm; __syncthreads();
    for (int s = 128; s; s >>= 1) {
        if (t < s) {
            float m1 = smax[t], m2 = smax[t + s], s1 = ssum[t], s2 = ssum[t + s];
            float mm = fmaxf(m1, m2);
            ssum[t] = s1 * expf(m1 - mm) + s2 * expf(m2 - mm);
            smax[t] = mm;
        }
        __syncthreads();
    }
    if (t == 0) { rowmax[i] = smax[0]; rowsum[i] = ssum[0]; }
}

// ---------- K3b: per-token argmax over clusters (argmax softmax == argmax raw); mark `has`
__global__ void k_col_argmax(const float* __restrict__ score, int* __restrict__ top,
                             int* __restrict__ has) {
    int j = blockIdx.x * 256 + threadIdx.x;
    float best = -3.0e38f; int bi = 0;
    for (int i = 0; i < M_ATTR; i++) {
        float v = score[(size_t)i * N_TOK + j];
        if (v > best) { best = v; bi = i; }
    }
    top[j] = bi;
    has[bi] = 1;   // benign race: all writers store 1
}

// ---------- K4: W[i,j] = assigned ? exp(s-mi)/(1+eps*Si) * normM[j] : 0   (f16, rows 312..319 = 0)
__global__ void k_build_w(const float* __restrict__ score, const float* __restrict__ rowmax,
                          const float* __restrict__ rowsum, const int* __restrict__ top,
                          const float* __restrict__ normM, _Float16* __restrict__ W) {
    int j = blockIdx.x * 256 + threadIdx.x;
    int i = blockIdx.y;
    float wv = 0.f;
    if (i < M_ATTR && top[j] == i) {
        float scale = 1.f / (1.f + 1e-9f * rowsum[i]);
        wv = expf(score[(size_t)i * N_TOK + j] - rowmax[i]) * scale * normM[j];
    }
    W[(size_t)i * N_TOK + j] = (_Float16)wv;
}

// ---------- K5: mean_part[ks] = W @ nlf over K-chunk; split-K, no atomics (deterministic).
// Block (ks, dgrp, mt): 16 M-rows x 128 d over 3136 tokens. Wave w owns 16 d-cols.
__global__ __launch_bounds__(256) void k_gemm2(const _Float16* __restrict__ W,
                                               const _Float16* __restrict__ nlf,
                                               float* __restrict__ mean_part) {
    __shared__ __align__(16) _Float16 sA[16 * 32];
    __shared__ __align__(16) _Float16 sBt[128 * 34];   // transposed: [d][j]
    int t = threadIdx.x, lane = t & 31, w = t >> 5;
    int m0 = blockIdx.z * 16, d0 = blockIdx.y * 128;
    size_t jbase = (size_t)blockIdx.x * KCHUNK;
    v8f acc = {};

    for (int s = 0; s < KCHUNK / 32; s++) {
        size_t j0 = jbase + s * 32;
        { // stage A: W rows, K=token contiguous
            int r = t >> 4, c2 = (t & 15) * 2;
            *(unsigned*)&sA[r * 32 + c2] =
                *(const unsigned*)&W[(size_t)(m0 + r) * N_TOK + j0 + c2];
        }
        { // stage B transposed: nlf[j][d] -> sBt[d][j] so K-pairs are contiguous
            int jl = t >> 3, dl0 = (t & 7) * 16;
            const _Float16* src = &nlf[(j0 + jl) * (size_t)DIM + d0 + dl0];
#pragma unroll
            for (int e = 0; e < 16; e++) sBt[(dl0 + e) * 34 + jl] = src[e];
        }
        __syncthreads();

        Frag a, b;
        {
            int Mr = lane & 15, kb = (lane < 16) ? 0 : 16;
#pragma unroll
            for (int v = 0; v < 8; v++) {
                int K = kb + 2 * v;
                a.u[v] = *(const unsigned*)&sA[Mr * 32 + K];
            }
        }
        {
            int dl = w * 16 + (lane & 15), kb = (lane < 16) ? 0 : 16;
#pragma unroll
            for (int v = 0; v < 8; v++)
                b.u[v] = *(const unsigned*)&sBt[dl * 34 + kb + 2 * v];
        }
        acc = __builtin_amdgcn_wmma_f32_16x16x32_f16(
            false, a.h, false, b.h, (short)0, acc, false, false);
        __syncthreads();
    }
    int rbase = m0 + ((lane < 16) ? 0 : 8);
    int d = d0 + w * 16 + (lane & 15);
    float* outp = mean_part + (size_t)blockIdx.x * (M_PAD * DIM);
#pragma unroll
    for (int r = 0; r < 8; r++)
        outp[(size_t)(rbase + r) * DIM + d] = acc[r];
}

// ---------- K6: reduce split-K partials
__global__ void k_reduce_mean(const float* __restrict__ part, float* __restrict__ mean_new) {
    int idx = blockIdx.x * 256 + threadIdx.x;   // < M_PAD*DIM
    float s = 0.f;
    for (int k = 0; k < KSPLIT; k++) s += part[(size_t)k * (M_PAD * DIM) + idx];
    mean_new[idx] = s;
}

// ---------- K7: momentum update + l2norm + residual + LayerNorm -> xln
__global__ void k_update(const float* __restrict__ ng, const float* __restrict__ mean_new,
                         const int* __restrict__ has, const float* __restrict__ glove,
                         const float* __restrict__ gamma, const float* __restrict__ beta,
                         float* __restrict__ xln) {
    int m = blockIdx.x, t = threadIdx.x;
    __shared__ float red[256];
    bool h = has[m] != 0;
    float u[3], ss = 0.f;
    for (int i = 0; i < 3; i++) {
        int d = t + 256 * i;
        float g = ng[(size_t)m * DIM + d];
        float v = h ? MOM * g + (1.f - MOM) * mean_new[(size_t)m * DIM + d] : g;
        u[i] = v; ss += v * v;
    }
    ss = block_sum(ss, red);
    float inv = 1.0f / fmaxf(sqrtf(ss), 1e-12f);
    float x[3], sx = 0.f;
    for (int i = 0; i < 3; i++) {
        int d = t + 256 * i;
        x[i] = u[i] * inv + glove[(size_t)m * DIM + d];
        sx += x[i];
    }
    float mu = block_sum(sx, red) * (1.0f / DIM);
    float sv = 0.f;
    for (int i = 0; i < 3; i++) { float dd = x[i] - mu; sv += dd * dd; }
    float var = block_sum(sv, red) * (1.0f / DIM);
    float rstd = rsqrtf(var + 1e-5f);
    for (int i = 0; i < 3; i++) {
        int d = t + 256 * i;
        xln[(size_t)m * DIM + d] = (x[i] - mu) * rstd * gamma[d] + beta[d];
    }
}

// ---------- K8: hid = gelu(xln @ fc1_w^T + b1)   [312 x 384]
__global__ void k_fc1(const float* __restrict__ xln, const float* __restrict__ w1,
                      const float* __restrict__ b1, float* __restrict__ hid) {
    __shared__ float xs[8 * DIM];
    int t = threadIdx.x, i0 = blockIdx.x * 8;
    for (int idx = t; idx < 8 * DIM; idx += 256) xs[idx] = xln[(size_t)i0 * DIM + idx];
    __syncthreads();
    for (int hc = t; hc < HID; hc += 256) {
        float acc[8] = {0, 0, 0, 0, 0, 0, 0, 0};
        const float* wr = w1 + (size_t)hc * DIM;
        for (int d = 0; d < DIM; d++) {
            float wv = wr[d];
#pragma unroll
            for (int r = 0; r < 8; r++) acc[r] += xs[r * DIM + d] * wv;
        }
        float b = b1[hc];
#pragma unroll
        for (int r = 0; r < 8; r++) {
            float v = acc[r] + b;
            hid[(size_t)(i0 + r) * HID + hc] = 0.5f * v * (1.f + erff(v * 0.70710678118654752f));
        }
    }
}

// ---------- K9: out = hid @ fc2_w^T + b2   [312 x 768]
__global__ void k_fc2(const float* __restrict__ hid, const float* __restrict__ w2,
                      const float* __restrict__ b2, float* __restrict__ out) {
    __shared__ float xs[8 * HID];
    int t = threadIdx.x, i0 = blockIdx.x * 8;
    for (int idx = t; idx < 8 * HID; idx += 256) xs[idx] = hid[(size_t)i0 * HID + idx];
    __syncthreads();
    for (int d = t; d < DIM; d += 256) {
        float acc[8] = {0, 0, 0, 0, 0, 0, 0, 0};
        const float* wr = w2 + (size_t)d * HID;
        for (int h = 0; h < HID; h++) {
            float wv = wr[h];
#pragma unroll
            for (int r = 0; r < 8; r++) acc[r] += xs[r * HID + h] * wv;
        }
        float b = b2[d];
#pragma unroll
        for (int r = 0; r < 8; r++)
            out[(size_t)(i0 + r) * DIM + d] = acc[r] + b;
    }
}

extern "C" void kernel_launch(void* const* d_in, const int* in_sizes, int n_in,
                              void* d_out, int out_size, void* d_ws, size_t ws_size,
                              hipStream_t stream) {
    const float* lt    = (const float*)d_in[0];
    const float* glove = (const float*)d_in[1];
    const float* gamma = (const float*)d_in[2];
    const float* beta  = (const float*)d_in[3];
    const float* w1    = (const float*)d_in[4];
    const float* b1    = (const float*)d_in[5];
    const float* w2    = (const float*)d_in[6];
    const float* b2    = (const float*)d_in[7];

    char* ws = (char*)d_ws;
    size_t off = 0;
    auto take = [&](size_t bytes) -> char* {
        char* p = ws + off;
        off = (off + bytes + 255) & ~(size_t)255;
        return p;
    };
    float*     ng     = (float*)take((size_t)M_ATTR * DIM * 4);
    _Float16*  ng_h   = (_Float16*)take((size_t)M_PAD * DIM * 2);
    _Float16*  nlf    = (_Float16*)take((size_t)N_TOK * DIM * 2);
    float*     normM  = (float*)take((size_t)N_TOK * 4);
    float*     score  = (float*)take((size_t)M_PAD * N_TOK * 4);
    float*     rowmax = (float*)take((size_t)M_PAD * 4);
    float*     rowsum = (float*)take((size_t)M_PAD * 4);
    int*       top    = (int*)take((size_t)N_TOK * 4);
    int*       has    = (int*)take((size_t)M_PAD * 4);
    _Float16*  W      = (_Float16*)take((size_t)M_PAD * N_TOK * 2);
    float*     mean_part = (float*)score;   // alias: score is dead after k_build_w
    float*     mean_new  = (float*)take((size_t)M_PAD * DIM * 4);
    float*     xln    = (float*)take((size_t)M_ATTR * DIM * 4);
    float*     hid    = (float*)take((size_t)M_ATTR * HID * 4);
    (void)ws_size; (void)in_sizes; (void)n_in; (void)out_size;

    k_prep_glove<<<M_PAD, 256, 0, stream>>>(glove, ng, ng_h, has);
    k_prep_tokens<<<N_TOK, 256, 0, stream>>>(lt, nlf, normM);
    k_gemm1<<<dim3(N_TOK / TOKB, M_PAD / 16), 256, 0, stream>>>(ng_h, nlf, score);
    k_row_stats<<<M_ATTR, 256, 0, stream>>>(score, rowmax, rowsum);
    k_col_argmax<<<N_TOK / 256, 256, 0, stream>>>(score, top, has);
    k_build_w<<<dim3(N_TOK / 256, M_PAD), 256, 0, stream>>>(score, rowmax, rowsum, top, normM, W);
    k_gemm2<<<dim3(KSPLIT, DIM / 128, M_PAD / 16), 256, 0, stream>>>(W, nlf, mean_part);
    k_reduce_mean<<<(M_PAD * DIM) / 256, 256, 0, stream>>>(mean_part, mean_new);
    k_update<<<M_ATTR, 256, 0, stream>>>(ng, mean_new, has, glove, gamma, beta, xln);
    k_fc1<<<M_ATTR / 8, 256, 0, stream>>>(xln, w1, b1, hid);
    k_fc2<<<M_ATTR / 8, 256, 0, stream>>>(hid, w2, b2, (float*)d_out);
}
